// SparseMoELayer_75926431858731
// MI455X (gfx1250) — compile-verified, matrix operand order checked by reading, not run on hardware
//
#include <hip/hip_runtime.h>
#include <math.h>

// Problem constants (match reference)
constexpr int Ntok = 16384;
constexpr int Dd   = 1024;     // input/output dim
constexpr int Hh   = 4096;     // expert hidden dim
constexpr int Ne   = 8;        // experts
constexpr int TopK = 2;

typedef __attribute__((ext_vector_type(16))) __bf16 v16bf;
typedef __attribute__((ext_vector_type(8)))  float  v8f;
typedef __attribute__((ext_vector_type(4)))  int    v4i;

// ---------- helpers ----------
__device__ __forceinline__ __bf16 f2bf(float f) {
    union { float f; unsigned u; } in; in.f = f;
    unsigned u = in.u;
    u += 0x7FFFu + ((u >> 16) & 1u);        // round-to-nearest-even
    union { unsigned short s; __bf16 b; } out;
    out.s = (unsigned short)(u >> 16);
    return out.b;
}

__device__ __forceinline__ float gelu_exact(float v) {
    return 0.5f * v * (1.0f + erff(v * 0.70710678118654752f));
}

// ---------- CDNA5 async global->LDS staging (guarded) ----------
#if __has_builtin(__builtin_amdgcn_global_load_async_to_lds_b128)
#define ASYNC_LDS 1
typedef __attribute__((address_space(1))) v4i gv4i;   // global int4
typedef __attribute__((address_space(3))) v4i lv4i;   // LDS int4
__device__ __forceinline__ void async_cp16(const void* g, void* l) {
    __builtin_amdgcn_global_load_async_to_lds_b128((gv4i*)g, (lv4i*)l, 0, 0);
}
__device__ __forceinline__ void async_wait0() {
#if __has_builtin(__builtin_amdgcn_s_wait_asynccnt)
    __builtin_amdgcn_s_wait_asynccnt(0);
#else
    asm volatile("s_wait_asynccnt 0x0" ::: "memory");
#endif
}
#endif

// copy 32 bytes (16 bf16) global -> LDS
__device__ __forceinline__ void cp32B(const __bf16* g, __bf16* l) {
#ifdef ASYNC_LDS
    async_cp16((const void*)g, (void*)l);
    async_cp16((const void*)(g + 8), (void*)(l + 8));
#else
    *(uint4*)l       = *(const uint4*)g;
    *(uint4*)(l + 8) = *(const uint4*)(g + 8);
#endif
}

// wait for my async copies + workgroup barrier
__device__ __forceinline__ void stage_fence() {
#ifdef ASYNC_LDS
    async_wait0();
#endif
    __syncthreads();
}

// ---------- 0: zero out + counts ----------
__global__ __launch_bounds__(256) void init_kernel(float* __restrict__ out,
                                                   int* __restrict__ counts) {
    size_t i = (size_t)blockIdx.x * blockDim.x + threadIdx.x;
    float4* o4 = (float4*)out;
    if (i < (size_t)Ntok * Dd / 4) o4[i] = make_float4(0.f, 0.f, 0.f, 0.f);
    if (i < Ne) counts[i] = 0;
}

// ---------- 1a: x fp32 -> bf16 ----------
__global__ __launch_bounds__(256) void xconv_kernel(const float* __restrict__ x,
                                                    __bf16* __restrict__ xb) {
    size_t i = ((size_t)blockIdx.x * blockDim.x + threadIdx.x) * 4;
    const float4 v = *(const float4*)(x + i);
    xb[i + 0] = f2bf(v.x); xb[i + 1] = f2bf(v.y);
    xb[i + 2] = f2bf(v.z); xb[i + 3] = f2bf(v.w);
}

// ---------- 1b: weight fp32 [E][R][C] -> bf16 transposed [E][C][R] ----------
__global__ __launch_bounds__(256) void transconv_kernel(
    const float* __restrict__ src, __bf16* __restrict__ dst, int R, int C) {
    __shared__ __bf16 tile[32][33];
    const size_t eoff = (size_t)blockIdx.z * R * C;
    const int c0 = blockIdx.x * 32, r0 = blockIdx.y * 32;
    const int tx = threadIdx.x & 31, ty = threadIdx.x >> 5;   // ty: 0..7
#pragma unroll
    for (int i = 0; i < 4; ++i)
        tile[ty + i * 8][tx] =
            f2bf(src[eoff + (size_t)(r0 + ty + i * 8) * C + c0 + tx]);
    __syncthreads();
#pragma unroll
    for (int i = 0; i < 4; ++i)
        dst[eoff + (size_t)(c0 + ty + i * 8) * R + r0 + tx] = tile[tx][ty + i * 8];
}

// ---------- 2: router (one wave32 per token) ----------
__global__ __launch_bounds__(256) void router_kernel(
    const float* __restrict__ x, const float* __restrict__ grad,
    const float* __restrict__ rW, const float* __restrict__ rb,
    float* __restrict__ probs_out, int* __restrict__ counts,
    int* __restrict__ tlist, float* __restrict__ glist) {
    const int wid  = (int)((blockIdx.x * (size_t)blockDim.x + threadIdx.x) >> 5);
    const int lane = threadIdx.x & 31;
    if (wid >= Ntok) return;
    const float* xr = x + (size_t)wid * Dd;

    float acc[Ne];
#pragma unroll
    for (int e = 0; e < Ne; ++e) acc[e] = 0.f;

    for (int d = lane; d < Dd; d += 32) {
        const float xv = xr[d];
#pragma unroll
        for (int e = 0; e < Ne; ++e) acc[e] += xv * rW[d * Ne + e];
    }
#pragma unroll
    for (int e = 0; e < Ne; ++e) {
#pragma unroll
        for (int off = 16; off; off >>= 1)
            acc[e] += __shfl_xor(acc[e], off, 32);
    }
    if (lane == 0) {
        const float g = grad[wid];
        float p[Ne];
        float mx = -3.0e38f;
#pragma unroll
        for (int e = 0; e < Ne; ++e) {
            p[e] = acc[e] + g * rW[Dd * Ne + e] + rb[e];
            mx = fmaxf(mx, p[e]);
        }
        float s = 0.f;
#pragma unroll
        for (int e = 0; e < Ne; ++e) { p[e] = expf(p[e] - mx); s += p[e]; }
        const float inv = 1.0f / s;
#pragma unroll
        for (int e = 0; e < Ne; ++e) {
            p[e] *= inv;
            probs_out[(size_t)wid * Ne + e] = p[e];
        }
        int i1 = 0;
#pragma unroll
        for (int e = 1; e < Ne; ++e) if (p[e] > p[i1]) i1 = e;
        int i2 = (i1 == 0) ? 1 : 0;
#pragma unroll
        for (int e = 0; e < Ne; ++e)
            if (e != i1 && p[e] > p[i2]) i2 = e;

        int pos = atomicAdd(&counts[i1], 1);
        tlist[i1 * Ntok + pos] = wid; glist[i1 * Ntok + pos] = p[i1];
        pos = atomicAdd(&counts[i2], 1);
        tlist[i2 * Ntok + pos] = wid; glist[i2 * Ntok + pos] = p[i2];
    }
}

// ---------- 3: exclusive prefix over 8 counts ----------
__global__ void offsets_kernel(const int* __restrict__ counts,
                               int* __restrict__ offs) {
    if (threadIdx.x == 0) {
        int s = 0;
        for (int e = 0; e < Ne; ++e) { offs[e] = s; s += counts[e]; }
    }
}

// ---------- tiled WMMA GEMMs ----------
// Block tile 128(m) x 128(n) x 32(k), 256 threads = 8 waves as 4(m) x 2(n).
// Wave tile 32x64: 2 A-frags x 4 B-frags -> 8 v_wmma_f32_16x16x32_bf16/k-step.
// Double-buffered LDS: async DMA of tile i+1 overlaps WMMA of tile i.
#define BM 128
#define BN 128
#define BK 32

// GEMM1: Hmat = gelu(gather(xb) @ W1T' + b1)   (W1T: [H][D] bf16)
__global__ __launch_bounds__(256) void gemm1_kernel(
    const __bf16* __restrict__ xb, const __bf16* __restrict__ W1T,
    const float* __restrict__ b1, const int* __restrict__ counts,
    const int* __restrict__ offs, const int* __restrict__ tlist,
    __bf16* __restrict__ Hmat) {
    const int e   = blockIdx.z;
    const int cnt = counts[e];
    const int m0  = blockIdx.x * BM;
    if (m0 >= cnt) return;
    const int n0  = blockIdx.y * BN;               // over Hh
    const int off = offs[e];

    __shared__ __align__(32) __bf16 As[2][BM * BK];
    __shared__ __align__(32) __bf16 Bs[2][BN * BK];

    const int tid  = threadIdx.x;
    const int lane = tid & 31;
    const int w    = tid >> 5;
    const int wm   = w >> 1;                       // 0..3
    const int wn   = w & 1;                        // 0..1

    // staging: thread t -> 16 consecutive bf16 of row (t>>1), kofs (t&1)*16
    const int srow = tid >> 1;
    const int kofs = (tid & 1) * 16;
    const int arow = m0 + srow;
    const int tok  = tlist[e * Ntok + ((arow < cnt) ? arow : m0)];
    const __bf16* srcA = xb + (size_t)tok * Dd + kofs;
    const __bf16* srcB = W1T + (size_t)e * Hh * Dd + (size_t)(n0 + srow) * Dd + kofs;

    const int rsel = lane & 15;
    const int ksel = (lane >> 4) * 16;
    int aoff[2], boff[4];
#pragma unroll
    for (int am = 0; am < 2; ++am)
        aoff[am] = (wm * 32 + am * 16 + rsel) * BK + ksel;
#pragma unroll
    for (int bn = 0; bn < 4; ++bn)
        boff[bn] = (wn * 64 + bn * 16 + rsel) * BK + ksel;

    v8f acc[2][4];
#pragma unroll
    for (int am = 0; am < 2; ++am)
#pragma unroll
        for (int bn = 0; bn < 4; ++bn) acc[am][bn] = (v8f){};

    constexpr int nIt = Dd / BK;
    cp32B(srcA, &As[0][tid * 16]);                 // prologue: tile 0 -> buf 0
    cp32B(srcB, &Bs[0][tid * 16]);

    for (int it = 0; it < nIt; ++it) {
        const int buf = it & 1;
        stage_fence();                             // buf ready; prior readers done
        if (it + 1 < nIt) {                        // DMA next tile into other buf
            cp32B(srcA + (it + 1) * BK, &As[buf ^ 1][tid * 16]);
            cp32B(srcB + (it + 1) * BK, &Bs[buf ^ 1][tid * 16]);
        }
        v16bf af[2], bf[4];
#pragma unroll
        for (int am = 0; am < 2; ++am) af[am] = *(const v16bf*)&As[buf][aoff[am]];
#pragma unroll
        for (int bn = 0; bn < 4; ++bn) bf[bn] = *(const v16bf*)&Bs[buf][boff[bn]];
#pragma unroll
        for (int am = 0; am < 2; ++am)
#pragma unroll
            for (int bn = 0; bn < 4; ++bn)
                acc[am][bn] = __builtin_amdgcn_wmma_f32_16x16x32_bf16(
                    false, af[am], false, bf[bn], (short)0, acc[am][bn],
                    false, false);
    }

    float bias[4];
    int   coln[4];
#pragma unroll
    for (int bn = 0; bn < 4; ++bn) {
        coln[bn] = n0 + wn * 64 + bn * 16 + (lane & 15);
        bias[bn] = b1[e * Hh + coln[bn]];
    }
#pragma unroll
    for (int am = 0; am < 2; ++am) {
        const int rbase = m0 + wm * 32 + am * 16 + ((lane >> 4) ? 8 : 0);
#pragma unroll
        for (int g = 0; g < 8; ++g) {
            const int r = rbase + g;
            if (r < cnt) {
#pragma unroll
                for (int bn = 0; bn < 4; ++bn)
                    Hmat[(size_t)(off + r) * Hh + coln[bn]] =
                        f2bf(gelu_exact(acc[am][bn][g] + bias[bn]));
            }
        }
    }
}

// GEMM2: out[token] += gate * (Hmat @ W2T' + b2)   (W2T: [D][H] bf16)
__global__ __launch_bounds__(256) void gemm2_kernel(
    const __bf16* __restrict__ Hmat, const __bf16* __restrict__ W2T,
    const float* __restrict__ b2, const int* __restrict__ counts,
    const int* __restrict__ offs, const int* __restrict__ tlist,
    const float* __restrict__ glist, float* __restrict__ out) {
    const int e   = blockIdx.z;
    const int cnt = counts[e];
    const int m0  = blockIdx.x * BM;
    if (m0 >= cnt) return;
    const int n0  = blockIdx.y * BN;               // over Dd
    const int off = offs[e];

    __shared__ __align__(32) __bf16 As[2][BM * BK];
    __shared__ __align__(32) __bf16 Bs[2][BN * BK];

    const int tid  = threadIdx.x;
    const int lane = tid & 31;
    const int w    = tid >> 5;
    const int wm   = w >> 1;
    const int wn   = w & 1;

    const int srow = tid >> 1;
    const int kofs = (tid & 1) * 16;
    const int arow = m0 + srow;
    const __bf16* srcA =
        Hmat + (size_t)(off + ((arow < cnt) ? arow : m0)) * Hh + kofs;
    const __bf16* srcB = W2T + (size_t)e * Dd * Hh + (size_t)(n0 + srow) * Hh + kofs;

    const int rsel = lane & 15;
    const int ksel = (lane >> 4) * 16;
    int aoff[2], boff[4];
#pragma unroll
    for (int am = 0; am < 2; ++am)
        aoff[am] = (wm * 32 + am * 16 + rsel) * BK + ksel;
#pragma unroll
    for (int bn = 0; bn < 4; ++bn)
        boff[bn] = (wn * 64 + bn * 16 + rsel) * BK + ksel;

    v8f acc[2][4];
#pragma unroll
    for (int am = 0; am < 2; ++am)
#pragma unroll
        for (int bn = 0; bn < 4; ++bn) acc[am][bn] = (v8f){};

    constexpr int nIt = Hh / BK;
    cp32B(srcA, &As[0][tid * 16]);
    cp32B(srcB, &Bs[0][tid * 16]);

    for (int it = 0; it < nIt; ++it) {
        const int buf = it & 1;
        stage_fence();
        if (it + 1 < nIt) {
            cp32B(srcA + (it + 1) * BK, &As[buf ^ 1][tid * 16]);
            cp32B(srcB + (it + 1) * BK, &Bs[buf ^ 1][tid * 16]);
        }
        v16bf af[2], bf[4];
#pragma unroll
        for (int am = 0; am < 2; ++am) af[am] = *(const v16bf*)&As[buf][aoff[am]];
#pragma unroll
        for (int bn = 0; bn < 4; ++bn) bf[bn] = *(const v16bf*)&Bs[buf][boff[bn]];
#pragma unroll
        for (int am = 0; am < 2; ++am)
#pragma unroll
            for (int bn = 0; bn < 4; ++bn)
                acc[am][bn] = __builtin_amdgcn_wmma_f32_16x16x32_bf16(
                    false, af[am], false, bf[bn], (short)0, acc[am][bn],
                    false, false);
    }

    float bias[4];
    int   coln[4];
#pragma unroll
    for (int bn = 0; bn < 4; ++bn) {
        coln[bn] = n0 + wn * 64 + bn * 16 + (lane & 15);
        bias[bn] = b2[e * Dd + coln[bn]];
    }
#pragma unroll
    for (int am = 0; am < 2; ++am) {
        const int rbase = m0 + wm * 32 + am * 16 + ((lane >> 4) ? 8 : 0);
#pragma unroll
        for (int g = 0; g < 8; ++g) {
            const int r = rbase + g;
            if (r < cnt) {
                const int   tok  = tlist[e * Ntok + r];
                const float gate = glist[e * Ntok + r];
#pragma unroll
                for (int bn = 0; bn < 4; ++bn)
                    atomicAdd(&out[(size_t)tok * Dd + coln[bn]],
                              gate * (acc[am][bn][g] + bias[bn]));
            }
        }
    }
}

// ---------- launch ----------
extern "C" void kernel_launch(void* const* d_in, const int* in_sizes, int n_in,
                              void* d_out, int out_size, void* d_ws, size_t ws_size,
                              hipStream_t stream) {
    const float* x    = (const float*)d_in[0];
    const float* grad = (const float*)d_in[1];
    const float* rW   = (const float*)d_in[2];
    const float* rb   = (const float*)d_in[3];
    const float* W1   = (const float*)d_in[4];
    const float* b1   = (const float*)d_in[5];
    const float* W2   = (const float*)d_in[6];
    const float* b2   = (const float*)d_in[7];

    float* out   = (float*)d_out;
    float* probs = out + (size_t)Ntok * Dd;

    // workspace partition
    char* p = (char*)d_ws;
    __bf16* W1T  = (__bf16*)p; p += (size_t)Ne * Hh * Dd * sizeof(__bf16);  // [E][H][D]
    __bf16* W2T  = (__bf16*)p; p += (size_t)Ne * Dd * Hh * sizeof(__bf16);  // [E][D][H]
    __bf16* Hmat = (__bf16*)p; p += (size_t)Ntok * TopK * Hh * sizeof(__bf16);
    __bf16* xb   = (__bf16*)p; p += (size_t)Ntok * Dd * sizeof(__bf16);
    int*   counts = (int*)p;   p += 256;
    int*   offs   = (int*)p;   p += 256;
    int*   tlist  = (int*)p;   p += (size_t)Ne * Ntok * sizeof(int);
    float* glist  = (float*)p; p += (size_t)Ne * Ntok * sizeof(float);

    // 0) zero out + counts
    init_kernel<<<(Ntok * Dd / 4 + 255) / 256, 256, 0, stream>>>(out, counts);
    // 1) x -> bf16 ; weights -> bf16 transposed
    xconv_kernel<<<((size_t)Ntok * Dd) / (256 * 4), 256, 0, stream>>>(x, xb);
    transconv_kernel<<<dim3(Hh / 32, Dd / 32, Ne), 256, 0, stream>>>(W1, W1T, Dd, Hh);
    transconv_kernel<<<dim3(Dd / 32, Hh / 32, Ne), 256, 0, stream>>>(W2, W2T, Hh, Dd);
    // 2) router + compaction
    router_kernel<<<Ntok / 8, 256, 0, stream>>>(x, grad, rW, rb, probs, counts,
                                                tlist, glist);
    // 3) prefix offsets
    offsets_kernel<<<1, 32, 0, stream>>>(counts, offs);
    // 4) expert GEMM1
    gemm1_kernel<<<dim3(Ntok / BM, Hh / BN, Ne), 256, 0, stream>>>(
        xb, W1T, b1, counts, offs, tlist, Hmat);
    // 5) expert GEMM2 (gated scatter-add)
    gemm2_kernel<<<dim3(Ntok / BM, Dd / BN, Ne), 256, 0, stream>>>(
        Hmat, W2T, b2, counts, offs, tlist, glist, out);
}
